// MultiheadSelfAttention_85358180041410
// MI455X (gfx1250) — compile-verified, matrix operand order checked by reading, not run on hardware
//
#include <hip/hip_runtime.h>
#include <hip/hip_bf16.h>

typedef __attribute__((ext_vector_type(16))) __bf16 v16bf;
typedef __attribute__((ext_vector_type(8)))  float  v8f;
typedef __attribute__((ext_vector_type(4)))  int    v4i;

union FragAB { uint4 u[2]; v16bf v; };  // 32 bytes = 16 bf16

static __device__ __forceinline__ unsigned short f32_to_bf16(float f) {
    unsigned int u = __float_as_uint(f);
    u += 0x7FFFu + ((u >> 16) & 1u);   // round-to-nearest-even
    return (unsigned short)(u >> 16);
}

static __device__ __forceinline__ v8f vzero8() {
    v8f z;
#pragma unroll
    for (int i = 0; i < 8; ++i) z[i] = 0.0f;
    return z;
}

#define DIMK 1024
#define SEQN 2048
#define NHEAD 16
#define HDIM 64

// ---------------------------------------------------------------------------
// Async global->LDS copy path (CDNA5 ASYNCcnt). Guarded so we fall back to the
// synchronous VGPR round-trip if the builtin isn't declared by this toolchain.
// ---------------------------------------------------------------------------
#define AS1 __attribute__((address_space(1)))
#define AS3 __attribute__((address_space(3)))

#if defined(__gfx1250__) && __has_builtin(__builtin_amdgcn_global_load_async_to_lds_b128)
#define USE_ASYNC_LDS 1
#else
#define USE_ASYNC_LDS 0
#endif

static __device__ __forceinline__ void copy_b128_to_lds(void* lds_dst, const void* gsrc) {
#if USE_ASYNC_LDS
    // flat->as(1): identity for global addresses; flat->as(3): low 32 bits are
    // the LDS byte offset (LDS aperture keeps the offset in [31:0]).
    AS1 v4i* g = (AS1 v4i*)(unsigned long long)gsrc;
    AS3 v4i* l = (AS3 v4i*)(unsigned int)(unsigned long long)lds_dst;
    __builtin_amdgcn_global_load_async_to_lds_b128(g, l, 0, 0);
#else
    *(uint4*)lds_dst = *(const uint4*)gsrc;
#endif
}

static __device__ __forceinline__ void async_copy_wait() {
#if USE_ASYNC_LDS
#if __has_builtin(__builtin_amdgcn_s_wait_asynccnt)
    __builtin_amdgcn_s_wait_asynccnt(0);
#else
    asm volatile("s_wait_asynccnt 0x0" ::: "memory");
#endif
#endif
}

// ---------------------------------------------------------------------------
// fp32 -> bf16 conversion
// ---------------------------------------------------------------------------
__global__ void cvt_bf16_kernel(const float* __restrict__ in,
                                unsigned short* __restrict__ out, int n) {
    int i = blockIdx.x * blockDim.x + threadIdx.x;
    if (i < n) out[i] = f32_to_bf16(in[i]);
}

// ---------------------------------------------------------------------------
// C[m,n] = sum_k A[m,k] * W[n,k] + bias[n]     (A,W bf16 row-major, K=N=1024)
// mode 0: store bf16 to [BH][2048][64]   (Q / K layout)
// mode 1: store bf16 to [BH][64][2048]   (V transposed layout)
// mode 2: store fp32 to [m][1024]        (final output)
// Block tile 128x128, 256 threads = 8 waves (4x2), wave tile 32x64.
// Double-buffered LDS; tile t+1 is DMA'd (async-to-LDS) while tile t is in WMMA.
// ---------------------------------------------------------------------------
__global__ __launch_bounds__(256)
void gemm_bf16_kernel(const unsigned short* __restrict__ A,
                      const unsigned short* __restrict__ W,
                      const float* __restrict__ bias,
                      void* __restrict__ out, int mode) {
    __shared__ unsigned short As[2][128][40];  // 32 + 8 pad (keeps 16B alignment)
    __shared__ unsigned short Bs[2][128][40];

    const int tid  = threadIdx.x;
    const int lane = tid & 31;
    const int wave = tid >> 5;
    const int wm   = wave >> 1;       // 0..3
    const int wn   = wave & 1;        // 0..1
    const int bm0  = blockIdx.y * 128;
    const int bn0  = blockIdx.x * 128;
    const int hl   = lane >> 4;       // half: 0/1
    const int l16  = lane & 15;
    const int kb   = hl * 8;          // kbase per ISA A/B layout

    // cooperative tile issue: 128 rows x 32 bf16 = 512 x b128 per matrix
    const int row0 = tid >> 2;              // thread's first row (i=0)
    const int c4   = tid & 3;
    auto issue_tile = [&](int k0, int st) {
#pragma unroll
        for (int i = 0; i < 2; ++i) {
            int row = row0 + i * 64;
            copy_b128_to_lds(&As[st][row][c4 * 8],
                             A + (size_t)(bm0 + row) * DIMK + k0 + c4 * 8);
            copy_b128_to_lds(&Bs[st][row][c4 * 8],
                             W + (size_t)(bn0 + row) * DIMK + k0 + c4 * 8);
        }
    };

    v8f acc[2][4];
#pragma unroll
    for (int mt = 0; mt < 2; ++mt)
#pragma unroll
        for (int nt = 0; nt < 4; ++nt) acc[mt][nt] = vzero8();

    // prologue: stage tile 0
    issue_tile(0, 0);
    async_copy_wait();
    __syncthreads();

    for (int k0 = 0; k0 < DIMK; k0 += 32) {
        const int st = (k0 >> 5) & 1;
        if (k0 + 32 < DIMK) issue_tile(k0 + 32, st ^ 1);  // DMA next tile

        FragAB afr[2], bfr[4];
#pragma unroll
        for (int mt = 0; mt < 2; ++mt) {
            int m = wm * 32 + mt * 16 + l16;
            afr[mt].u[0] = *(const uint4*)&As[st][m][kb];
            afr[mt].u[1] = *(const uint4*)&As[st][m][kb + 16];
        }
#pragma unroll
        for (int nt = 0; nt < 4; ++nt) {
            int n = wn * 64 + nt * 16 + l16;
            bfr[nt].u[0] = *(const uint4*)&Bs[st][n][kb];
            bfr[nt].u[1] = *(const uint4*)&Bs[st][n][kb + 16];
        }
#pragma unroll
        for (int mt = 0; mt < 2; ++mt)
#pragma unroll
            for (int nt = 0; nt < 4; ++nt)
                acc[mt][nt] = __builtin_amdgcn_wmma_f32_16x16x32_bf16(
                    false, afr[mt].v, false, bfr[nt].v,
                    (short)0, acc[mt][nt], false, false);

        async_copy_wait();   // next tile resident before anyone reads it
        __syncthreads();     // and all waves done reading current tile
    }

    // epilogue
#pragma unroll
    for (int mt = 0; mt < 2; ++mt) {
#pragma unroll
        for (int nt = 0; nt < 4; ++nt) {
#pragma unroll
            for (int r = 0; r < 8; ++r) {
                int m = bm0 + wm * 32 + mt * 16 + hl * 8 + r;
                int n = bn0 + wn * 64 + nt * 16 + l16;
                float v = acc[mt][nt][r] + bias[n];
                if (mode == 2) {
                    ((float*)out)[(size_t)m * DIMK + n] = v;
                } else {
                    int b  = m >> 11;            // / 2048
                    int nq = m & (SEQN - 1);
                    int h  = n >> 6;
                    int d  = n & (HDIM - 1);
                    int bh = b * NHEAD + h;
                    size_t di = (mode == 0)
                        ? ((size_t)bh * SEQN + nq) * HDIM + d
                        : ((size_t)bh * HDIM + d) * SEQN + nq;
                    ((unsigned short*)out)[di] = f32_to_bf16(v);
                }
            }
        }
    }
}

// ---------------------------------------------------------------------------
// Flash attention: one (head, 64-query-block) per workgroup. 128 thr = 4 waves,
// each wave owns 16 query rows. Online softmax, scores never touch HBM.
// Q,K: [BH][2048][64] bf16.  Vt: [BH][64][2048] bf16.  Out bf16 [B][2048][1024]
// ---------------------------------------------------------------------------
__global__ __launch_bounds__(128)
void attn_kernel(const unsigned short* __restrict__ Q,
                 const unsigned short* __restrict__ K,
                 const unsigned short* __restrict__ Vt,
                 const unsigned char* __restrict__ mask,
                 unsigned short* __restrict__ Ob) {
    __shared__ unsigned short Ps[4][16][72];   // per-wave P staging (C->A layout)

    const int bh   = blockIdx.x;          // 0..31
    const int qb   = blockIdx.y * 64;
    const int b    = bh >> 4;
    const int tid  = threadIdx.x;
    const int lane = tid & 31;
    const int wave = tid >> 5;
    const int hl   = lane >> 4;
    const int l16  = lane & 15;
    const int kb   = hl * 8;

    const unsigned short* Qh = Q  + (size_t)bh * SEQN * HDIM;
    const unsigned short* Kh = K  + (size_t)bh * SEQN * HDIM;
    const unsigned short* Vh = Vt + (size_t)bh * HDIM * SEQN;

    // preload this wave's Q fragments (16 rows x 64 d, two K-chunks of 32)
    FragAB qfr[2];
    {
        const unsigned short* qrow = Qh + (size_t)(qb + wave * 16 + l16) * HDIM;
#pragma unroll
        for (int kc = 0; kc < 2; ++kc) {
            qfr[kc].u[0] = *(const uint4*)(qrow + kc * 32 + kb);
            qfr[kc].u[1] = *(const uint4*)(qrow + kc * 32 + kb + 16);
        }
    }

    float mrow[8], lrow[8];
    v8f accO[4];
#pragma unroll
    for (int r = 0; r < 8; ++r) { mrow[r] = -1e30f; lrow[r] = 0.0f; }
#pragma unroll
    for (int nt = 0; nt < 4; ++nt) accO[nt] = vzero8();

    for (int kv0 = 0; kv0 < SEQN; kv0 += 64) {
        // ---- S = Q K^T  (4 N-tiles of 16 keys) ----
        v8f s[4];
#pragma unroll
        for (int nt = 0; nt < 4; ++nt) s[nt] = vzero8();
#pragma unroll
        for (int kc = 0; kc < 2; ++kc) {
#pragma unroll
            for (int nt = 0; nt < 4; ++nt) {
                FragAB kf;
                const unsigned short* krow =
                    Kh + (size_t)(kv0 + nt * 16 + l16) * HDIM + kc * 32;
                kf.u[0] = *(const uint4*)(krow + kb);
                kf.u[1] = *(const uint4*)(krow + kb + 16);
                s[nt] = __builtin_amdgcn_wmma_f32_16x16x32_bf16(
                    false, qfr[kc].v, false, kf.v, (short)0, s[nt], false, false);
            }
        }
        // ---- scale + mask ----
#pragma unroll
        for (int nt = 0; nt < 4; ++nt) {
            bool mk = mask[b * SEQN + kv0 + nt * 16 + l16] != 0;
#pragma unroll
            for (int r = 0; r < 8; ++r) {
                float v = s[nt][r] * 0.125f;
                s[nt][r] = mk ? -20000.0f : v;
            }
        }
        // ---- online softmax: per-row max / sum (rows split per C layout) ----
        float mnew[8], corr[8];
#pragma unroll
        for (int r = 0; r < 8; ++r) {
            float t = fmaxf(fmaxf(s[0][r], s[1][r]), fmaxf(s[2][r], s[3][r]));
#pragma unroll
            for (int off = 1; off < 16; off <<= 1)
                t = fmaxf(t, __shfl_xor(t, off, 32));
            mnew[r] = fmaxf(mrow[r], t);
            corr[r] = __expf(mrow[r] - mnew[r]);
            mrow[r] = mnew[r];
        }
#pragma unroll
        for (int r = 0; r < 8; ++r) {
            float sum = 0.0f;
#pragma unroll
            for (int nt = 0; nt < 4; ++nt) {
                float p = __expf(s[nt][r] - mnew[r]);
                s[nt][r] = p;
                sum += p;
            }
#pragma unroll
            for (int off = 1; off < 16; off <<= 1)
                sum += __shfl_xor(sum, off, 32);
            lrow[r] = lrow[r] * corr[r] + sum;
        }
#pragma unroll
        for (int nt = 0; nt < 4; ++nt)
#pragma unroll
            for (int r = 0; r < 8; ++r) accO[nt][r] *= corr[r];

        // ---- P: C layout -> A layout via per-wave LDS buffer ----
#pragma unroll
        for (int nt = 0; nt < 4; ++nt)
#pragma unroll
            for (int r = 0; r < 8; ++r)
                Ps[wave][hl * 8 + r][nt * 16 + l16] = f32_to_bf16(s[nt][r]);
        // same-wave LDS RAW: hardware keeps DS ops in order; compiler waits DScnt

        // ---- O += P V  (Vt rows are contiguous over keys) ----
#pragma unroll
        for (int kc = 0; kc < 2; ++kc) {
            FragAB pf;
            pf.u[0] = *(const uint4*)&Ps[wave][l16][kc * 32 + kb];
            pf.u[1] = *(const uint4*)&Ps[wave][l16][kc * 32 + kb + 16];
#pragma unroll
            for (int nt = 0; nt < 4; ++nt) {
                FragAB vf;
                const unsigned short* vrow =
                    Vh + (size_t)(nt * 16 + l16) * SEQN + kv0 + kc * 32;
                vf.u[0] = *(const uint4*)(vrow + kb);
                vf.u[1] = *(const uint4*)(vrow + kb + 16);
                accO[nt] = __builtin_amdgcn_wmma_f32_16x16x32_bf16(
                    false, pf.v, false, vf.v, (short)0, accO[nt], false, false);
            }
        }
    }

    // ---- normalize and store [B][N][H*64] bf16 ----
#pragma unroll
    for (int r = 0; r < 8; ++r) lrow[r] = 1.0f / lrow[r];
    const int h = bh & (NHEAD - 1);
#pragma unroll
    for (int nt = 0; nt < 4; ++nt) {
#pragma unroll
        for (int r = 0; r < 8; ++r) {
            int nq = qb + wave * 16 + hl * 8 + r;
            int d  = nt * 16 + l16;
            Ob[((size_t)b * SEQN + nq) * DIMK + h * HDIM + d] =
                f32_to_bf16(accO[nt][r] * lrow[r]);
        }
    }
}

// ---------------------------------------------------------------------------
extern "C" void kernel_launch(void* const* d_in, const int* in_sizes, int n_in,
                              void* d_out, int out_size, void* d_ws, size_t ws_size,
                              hipStream_t stream) {
    const float* x          = (const float*)d_in[0];
    const unsigned char* mk = (const unsigned char*)d_in[1];
    const float* Wq = (const float*)d_in[2];
    const float* bq = (const float*)d_in[3];
    const float* Wk = (const float*)d_in[4];
    const float* bk = (const float*)d_in[5];
    const float* Wv = (const float*)d_in[6];
    const float* bv = (const float*)d_in[7];
    const float* Wo = (const float*)d_in[8];
    const float* bo = (const float*)d_in[9];
    float* out = (float*)d_out;

    const size_t NX = (size_t)2 * SEQN * DIMK;   // 4,194,304
    const size_t NW = (size_t)DIMK * DIMK;       // 1,048,576

    unsigned short* ws  = (unsigned short*)d_ws;
    unsigned short* xb  = ws;
    unsigned short* wqb = xb  + NX;
    unsigned short* wkb = wqb + NW;
    unsigned short* wvb = wkb + NW;
    unsigned short* wob = wvb + NW;
    unsigned short* Qb  = wob + NW;
    unsigned short* Kb  = Qb  + NX;
    unsigned short* Vtb = Kb  + NX;
    unsigned short* Ab  = Vtb + NX;

    const int CT = 256;
    cvt_bf16_kernel<<<(int)((NX + CT - 1) / CT), CT, 0, stream>>>(x,  xb,  (int)NX);
    cvt_bf16_kernel<<<(int)((NW + CT - 1) / CT), CT, 0, stream>>>(Wq, wqb, (int)NW);
    cvt_bf16_kernel<<<(int)((NW + CT - 1) / CT), CT, 0, stream>>>(Wk, wkb, (int)NW);
    cvt_bf16_kernel<<<(int)((NW + CT - 1) / CT), CT, 0, stream>>>(Wv, wvb, (int)NW);
    cvt_bf16_kernel<<<(int)((NW + CT - 1) / CT), CT, 0, stream>>>(Wo, wob, (int)NW);

    dim3 ggrid(DIMK / 128, (2 * SEQN) / 128);   // (8, 32)
    gemm_bf16_kernel<<<ggrid, 256, 0, stream>>>(xb, wqb, bq, Qb,  0);
    gemm_bf16_kernel<<<ggrid, 256, 0, stream>>>(xb, wkb, bk, Kb,  0);
    gemm_bf16_kernel<<<ggrid, 256, 0, stream>>>(xb, wvb, bv, Vtb, 1);

    attn_kernel<<<dim3(2 * NHEAD, SEQN / 64), 128, 0, stream>>>(Qb, Kb, Vtb, mk, Ab);

    gemm_bf16_kernel<<<ggrid, 256, 0, stream>>>(Ab, wob, bo, out, 2);
}
